// LatentActionIDM_31911607009321
// MI455X (gfx1250) — compile-verified, matrix operand order checked by reading, not run on hardware
//
#include <hip/hip_runtime.h>
#include <math.h>

// ---------------------------------------------------------------------------
// Problem constants
// ---------------------------------------------------------------------------
#define Bdim 64
#define Tdim 128
#define Ddim 1024
#define Edim 2048
#define CDdim 256
#define NCdim 1024
#define K2dim (2 * Edim)             // 4096, GEMM2 reduction dim
#define Nrows (Bdim * Tdim)          // 8192 rows of latent
#define Mstack (2 * Nrows)           // 16384 stacked rows for GEMM1

// d_out layout (float32 elements), in reference return order:
// quantize_st (8192*256), loss (1), perplexity (1), encodings (8192*1024), indices (8192)
#define Q_OFF    ((size_t)0)
#define LOSS_OFF ((size_t)(Nrows * CDdim))
#define PERP_OFF (LOSS_OFF + 1)
#define ENC_OFF  (PERP_OFF + 1)
#define IDX_OFF  (ENC_OFF + (size_t)Nrows * NCdim)

typedef __attribute__((ext_vector_type(16))) __bf16 v16bf;
typedef __attribute__((ext_vector_type(8)))  __bf16 v8bf;
typedef __attribute__((ext_vector_type(4)))  __bf16 v4bf;
typedef __attribute__((ext_vector_type(8)))  float  v8f;
typedef __attribute__((ext_vector_type(4)))  float  v4f;
typedef __attribute__((ext_vector_type(4)))  unsigned int v4u;
typedef __attribute__((ext_vector_type(8)))  int    v8i;
typedef __attribute__((ext_vector_type(4)))  int    v4i;

__device__ __forceinline__ float gelu_tanh(float x) {
    // jax.nn.gelu (approximate=True)
    float x3 = x * x * x;
    return 0.5f * x * (1.0f + tanhf(0.7978845608028654f * (x + 0.044715f * x3)));
}

__device__ __forceinline__ v16bf join8(v8bf a0, v8bf a1) {
    return __builtin_shufflevector(a0, a1, 0, 1, 2, 3, 4, 5, 6, 7,
                                   8, 9, 10, 11, 12, 13, 14, 15);
}

// ---------------------------------------------------------------------------
// Tensor Data Mover: 2D bf16 tile (tile_d0 x tile_d1) from global -> LDS.
// D# bitfields per CDNA5 ISA §8.3/§8.4.  Dims/strides in data_size units,
// addresses in bytes.  Issued once per wave (EXEC-independent DMA).
// 6-arg builtin form: (g0, g1, g2, g3, g_extra, cpol).
// ---------------------------------------------------------------------------
__device__ __forceinline__ void tdm_load_2d_bf16(unsigned int lds_addr,
                                                 unsigned long long gaddr,
                                                 unsigned int tensor_d0,
                                                 unsigned int tensor_d1,
                                                 unsigned long long stride0,
                                                 unsigned int tile_d0,
                                                 unsigned int tile_d1) {
    v4u g0;
    g0.x = 1u;                                           // count=1, user D#
    g0.y = lds_addr;                                     // LDS byte address
    g0.z = (unsigned int)gaddr;                          // global_addr[31:0]
    g0.w = (unsigned int)((gaddr >> 32) & 0x01FFFFFFu)   // global_addr[56:32]
           | (2u << 30);                                 // type = 2 ("image")
    v8i g1;
    g1[0] = (int)(1u << 16);                             // data_size=1 -> 2 bytes
    g1[1] = (int)((tensor_d0 & 0xFFFFu) << 16);          // tensor_dim0[15:0]
    g1[2] = (int)(((tensor_d0 >> 16) & 0xFFFFu) |
                  ((tensor_d1 & 0xFFFFu) << 16));        // dim0[31:16] | dim1[15:0]
    g1[3] = (int)(((tensor_d1 >> 16) & 0xFFFFu) |
                  (tile_d0 << 16));                      // dim1[31:16] | tile_dim0
    g1[4] = (int)tile_d1;                                // tile_dim1; tile_dim2=0
    g1[5] = (int)(unsigned int)stride0;                  // dim0_stride[31:0]
    g1[6] = (int)((stride0 >> 32) & 0xFFFFu);            // dim0_stride[47:32]
    g1[7] = 0;
    v4i z4 = {0, 0, 0, 0};
    v8i z8 = {0, 0, 0, 0, 0, 0, 0, 0};
    __builtin_amdgcn_tensor_load_to_lds(g0, g1, z4, z4, z8, 0);
}

// ---------------------------------------------------------------------------
// fp32 -> bf16 bulk convert (4 elements per thread-iteration)
// ---------------------------------------------------------------------------
__global__ void cvt_bf16_kernel(const float* __restrict__ src, __bf16* __restrict__ dst,
                                size_t n4) {
    for (size_t e = (size_t)blockIdx.x * blockDim.x + threadIdx.x; e < n4;
         e += (size_t)gridDim.x * blockDim.x) {
        v4f v = *(const v4f*)&src[e * 4];
        *(v4bf*)&dst[e * 4] = __builtin_convertvector(v, v4bf);
    }
}

// ---------------------------------------------------------------------------
// Pack a (K x N) fp32 B matrix (row-major; or transpose => src is (N x K))
// into WMMA bf16 lane order: dst[((nTile*KC + kc)*32 + lane)*16 + i]
// ---------------------------------------------------------------------------
__global__ void pack_b_kernel(const float* __restrict__ src, __bf16* __restrict__ dst,
                              int K, int N, int transpose) {
    size_t total = (size_t)K * N;
    int KC = K >> 5;
    for (size_t e = (size_t)blockIdx.x * blockDim.x + threadIdx.x; e < total;
         e += (size_t)gridDim.x * blockDim.x) {
        int i    = (int)(e & 15);
        int lane = (int)((e >> 4) & 31);
        size_t rest = e >> 9;
        int kc    = (int)(rest % KC);
        int nTile = (int)(rest / KC);
        int k = kc * 32 + ((lane >> 4) * 8) + (i & 7) + ((i >> 3) * 16);
        int n = nTile * 16 + (lane & 15);
        float v = transpose ? src[(size_t)n * K + k] : src[(size_t)k * N + n];
        dst[e] = (__bf16)v;
    }
}

// ---------------------------------------------------------------------------
// Codebook squared norms (fp32 for a robust argmin)
// ---------------------------------------------------------------------------
__global__ void cnorm_kernel(const float* __restrict__ cb, float* __restrict__ cn) {
    __shared__ float sred[256];
    int n = blockIdx.x, t = threadIdx.x;
    float v = cb[(size_t)n * CDdim + t];
    sred[t] = v * v;
    __syncthreads();
    for (int s = 128; s > 0; s >>= 1) {
        if (t < s) sred[t] += sred[t + s];
        __syncthreads();
    }
    if (t == 0) cn[n] = sred[0];
}

// ---------------------------------------------------------------------------
// Zero init: encodings region of d_out, histogram counts, loss accumulator
// ---------------------------------------------------------------------------
__global__ void init_kernel(float* __restrict__ enc, size_t enc_n,
                            int* __restrict__ counts, float* __restrict__ lossAcc) {
    size_t tid = (size_t)blockIdx.x * blockDim.x + threadIdx.x;
    for (size_t e = tid; e < enc_n; e += (size_t)gridDim.x * blockDim.x) enc[e] = 0.0f;
    if (tid < NCdim) counts[tid] = 0;
    if (tid == 0) *lossAcc = 0.0f;
}

// ---------------------------------------------------------------------------
// GEMM1: policy = gelu(concat([S;NS] @ W_s + b_s))  (bf16 out)
// Block tile 64 rows x 128 cols; 8 waves; wave w owns N-tile (by*8+w) and
// 4 M-subtile accumulators.  A tiles (64x64 bf16) staged via TDM, double-
// buffered in LDS; 8 WMMAs per wave per barrier round.
// grid = (Mstack/64, Edim/128)
// ---------------------------------------------------------------------------
__global__ void gemm1_kernel(const __bf16* __restrict__ Abf,
                             const __bf16* __restrict__ Wspk,
                             const float* __restrict__ b_s,
                             __bf16* __restrict__ policy) {
    const int mBlock = blockIdx.x;            // 0..255 (64 rows each)
    const int w      = threadIdx.x >> 5;
    const int lane   = threadIdx.x & 31;
    const int nTile  = blockIdx.y * 8 + w;    // 0..127
    __shared__ __bf16 ldsA[2][64 * 64];       // 2 x 8KB stages

    const int row  = lane & 15;
    const int base = (lane >> 4) * 8;
    v8f acc[4] = {};

    const unsigned long long abase =
        (unsigned long long)(uintptr_t)Abf + (unsigned long long)mBlock * 64 * Ddim * 2;
    const int NSTAGE = Ddim / 64;             // 16

    if (w == 0) {
        tdm_load_2d_bf16((unsigned int)(uintptr_t)&ldsA[0][0], abase,
                         Ddim, Mstack, Ddim, 64, 64);
        __builtin_amdgcn_s_wait_tensorcnt(0);
    }
    __syncthreads();

    for (int ks = 0; ks < NSTAGE; ++ks) {
        const int cur = ks & 1;
        if (w == 0 && ks + 1 < NSTAGE) {
            tdm_load_2d_bf16((unsigned int)(uintptr_t)&ldsA[cur ^ 1][0],
                             abase + (unsigned long long)(ks + 1) * 64 * 2,
                             Ddim, Mstack, Ddim, 64, 64);
        }
#pragma unroll
        for (int s = 0; s < 2; ++s) {
            const int kg = ks * 2 + s;
            v16bf b = *(const v16bf*)&Wspk[(((size_t)nTile * (Ddim / 32) + kg) * 32 + lane) * 16];
#pragma unroll
            for (int ms = 0; ms < 4; ++ms) {
                const __bf16* ap = &ldsA[cur][(ms * 16 + row) * 64 + s * 32 + base];
                v16bf a = join8(*(const v8bf*)ap, *(const v8bf*)(ap + 16));
                acc[ms] = __builtin_amdgcn_wmma_f32_16x16x32_bf16(false, a, false, b,
                                                                  (short)0, acc[ms],
                                                                  false, false);
            }
        }
        if (w == 0) __builtin_amdgcn_s_wait_tensorcnt(0);
        __syncthreads();
    }

    const int colg = nTile * 16 + (lane & 15);
    const float bias = b_s[colg];
#pragma unroll
    for (int ms = 0; ms < 4; ++ms) {
#pragma unroll
        for (int r = 0; r < 8; ++r) {
            size_t gm = (size_t)mBlock * 64 + ms * 16 + r + 8 * (lane >> 4);
            float g = gelu_tanh(acc[ms][r] + bias);
            size_t prow = (gm < Nrows) ? gm : gm - Nrows;
            int nout = (gm < Nrows) ? colg : colg + Edim;
            policy[prow * K2dim + nout] = (__bf16)g;
        }
    }
}

// ---------------------------------------------------------------------------
// GEMM2: latent = policy @ W_p + b_p -> latent fp32 + bf16
// Same structure; K = 4096.  grid = (Nrows/64, CDdim/128)
// ---------------------------------------------------------------------------
__global__ void gemm2_kernel(const __bf16* __restrict__ policy,
                             const __bf16* __restrict__ Wppk,
                             const float* __restrict__ b_p,
                             float* __restrict__ latf,
                             __bf16* __restrict__ latbf) {
    const int mBlock = blockIdx.x;            // 0..127
    const int w      = threadIdx.x >> 5;
    const int lane   = threadIdx.x & 31;
    const int nTile  = blockIdx.y * 8 + w;    // 0..15
    __shared__ __bf16 ldsA[2][64 * 64];

    const int row  = lane & 15;
    const int base = (lane >> 4) * 8;
    v8f acc[4] = {};

    const unsigned long long abase =
        (unsigned long long)(uintptr_t)policy + (unsigned long long)mBlock * 64 * K2dim * 2;
    const int NSTAGE = K2dim / 64;            // 64

    if (w == 0) {
        tdm_load_2d_bf16((unsigned int)(uintptr_t)&ldsA[0][0], abase,
                         K2dim, Nrows, K2dim, 64, 64);
        __builtin_amdgcn_s_wait_tensorcnt(0);
    }
    __syncthreads();

    for (int ks = 0; ks < NSTAGE; ++ks) {
        const int cur = ks & 1;
        if (w == 0 && ks + 1 < NSTAGE) {
            tdm_load_2d_bf16((unsigned int)(uintptr_t)&ldsA[cur ^ 1][0],
                             abase + (unsigned long long)(ks + 1) * 64 * 2,
                             K2dim, Nrows, K2dim, 64, 64);
        }
#pragma unroll
        for (int s = 0; s < 2; ++s) {
            const int kg = ks * 2 + s;
            v16bf b = *(const v16bf*)&Wppk[(((size_t)nTile * (K2dim / 32) + kg) * 32 + lane) * 16];
#pragma unroll
            for (int ms = 0; ms < 4; ++ms) {
                const __bf16* ap = &ldsA[cur][(ms * 16 + row) * 64 + s * 32 + base];
                v16bf a = join8(*(const v8bf*)ap, *(const v8bf*)(ap + 16));
                acc[ms] = __builtin_amdgcn_wmma_f32_16x16x32_bf16(false, a, false, b,
                                                                  (short)0, acc[ms],
                                                                  false, false);
            }
        }
        if (w == 0) __builtin_amdgcn_s_wait_tensorcnt(0);
        __syncthreads();
    }

    const int colg = nTile * 16 + (lane & 15);
    const float bias = b_p[colg];
#pragma unroll
    for (int ms = 0; ms < 4; ++ms) {
#pragma unroll
        for (int r = 0; r < 8; ++r) {
            size_t gm = (size_t)mBlock * 64 + ms * 16 + r + 8 * (lane >> 4);
            float x = acc[ms][r] + bias;
            latf[gm * CDdim + colg]  = x;
            latbf[gm * CDdim + colg] = (__bf16)x;
        }
    }
}

// ---------------------------------------------------------------------------
// GEMM3 + argmin: argmin_c (cnorm[c] - 2 * latent . codebook[c])
// grid = Nrows/16; block = 256; wave w scans nTiles w*8..w*8+7.
// ---------------------------------------------------------------------------
__global__ void gemm3_argmin_kernel(const __bf16* __restrict__ latbf,
                                    const __bf16* __restrict__ CBpk,
                                    const float* __restrict__ cn,
                                    int* __restrict__ idxbuf) {
    const int mTile = blockIdx.x;             // 0..511
    const int w     = threadIdx.x >> 5;
    const int lane  = threadIdx.x & 31;
    __shared__ __bf16 ldsA[16 * CDdim];       // 8 KB
    __shared__ float redV[8][16];
    __shared__ int   redI[8][16];

    *(v16bf*)&ldsA[threadIdx.x * 16] =
        *(const v16bf*)&latbf[(size_t)mTile * 16 * CDdim + threadIdx.x * 16];
    __syncthreads();

    const int row  = lane & 15;
    const int base = (lane >> 4) * 8;

    float bestV[8];
    int   bestI[8];
#pragma unroll
    for (int r = 0; r < 8; ++r) { bestV[r] = 3.4e38f; bestI[r] = 0; }

    for (int j = 0; j < 8; ++j) {
        const int nTile = w * 8 + j;
        v8f acc = {};
#pragma unroll
        for (int kc = 0; kc < CDdim / 32; ++kc) {
            const __bf16* ap = &ldsA[row * CDdim + kc * 32 + base];
            v16bf a = join8(*(const v8bf*)ap, *(const v8bf*)(ap + 16));
            v16bf b = *(const v16bf*)&CBpk[(((size_t)nTile * (CDdim / 32) + kc) * 32 + lane) * 16];
            acc = __builtin_amdgcn_wmma_f32_16x16x32_bf16(false, a, false, b,
                                                          (short)0, acc, false, false);
        }
        const int n = nTile * 16 + (lane & 15);
        const float cnv = cn[n];
#pragma unroll
        for (int r = 0; r < 8; ++r) {
            float d = cnv - 2.0f * acc[r];
            if (d < bestV[r] || (d == bestV[r] && n < bestI[r])) { bestV[r] = d; bestI[r] = n; }
        }
    }

#pragma unroll
    for (int r = 0; r < 8; ++r) {
        for (int off = 8; off >= 1; off >>= 1) {
            float ov = __shfl_xor(bestV[r], off, 32);
            int   oi = __shfl_xor(bestI[r], off, 32);
            if (ov < bestV[r] || (ov == bestV[r] && oi < bestI[r])) { bestV[r] = ov; bestI[r] = oi; }
        }
    }
    if ((lane & 15) == 0) {
#pragma unroll
        for (int r = 0; r < 8; ++r) {
            int mi = r + 8 * (lane >> 4);
            redV[w][mi] = bestV[r];
            redI[w][mi] = bestI[r];
        }
    }
    __syncthreads();
    if (threadIdx.x < 16) {
        float bv = redV[0][threadIdx.x];
        int   bi = redI[0][threadIdx.x];
#pragma unroll
        for (int ww = 1; ww < 8; ++ww) {
            float ov = redV[ww][threadIdx.x];
            int   oi = redI[ww][threadIdx.x];
            if (ov < bv || (ov == bv && oi < bi)) { bv = ov; bi = oi; }
        }
        idxbuf[mTile * 16 + threadIdx.x] = bi;
    }
}

// ---------------------------------------------------------------------------
// Gather quantize = codebook[idx], one-hot, indices, histogram, loss accum.
// ---------------------------------------------------------------------------
__global__ void scatter_kernel(const int* __restrict__ idxbuf,
                               const float* __restrict__ codebook,
                               const float* __restrict__ latf,
                               float* __restrict__ out,
                               int* __restrict__ counts,
                               float* __restrict__ lossAcc) {
    __shared__ float sred[256];
    const int n = blockIdx.x;
    const int t = threadIdx.x;
    const int i = idxbuf[n];
    float q = codebook[(size_t)i * CDdim + t];
    out[Q_OFF + (size_t)n * CDdim + t] = q;   // quantize_st forward == quantize
    float diff = q - latf[(size_t)n * CDdim + t];
    sred[t] = diff * diff;
    __syncthreads();
    for (int s = 128; s > 0; s >>= 1) {
        if (t < s) sred[t] += sred[t + s];
        __syncthreads();
    }
    if (t == 0) {
        atomicAdd(lossAcc, sred[0]);
        out[ENC_OFF + (size_t)n * NCdim + i] = 1.0f;
        out[IDX_OFF + n] = (float)i;
        atomicAdd(&counts[i], 1);
    }
}

// ---------------------------------------------------------------------------
// Perplexity + loss finalize.
// ---------------------------------------------------------------------------
__global__ void finalize_kernel(const int* __restrict__ counts,
                                const float* __restrict__ lossAcc,
                                float* __restrict__ out) {
    __shared__ float sred[1024];
    int t = threadIdx.x;
    float p = (float)counts[t] * (1.0f / (float)Nrows);
    sred[t] = p * logf(p + 1e-10f);
    __syncthreads();
    for (int s = 512; s > 0; s >>= 1) {
        if (t < s) sred[t] += sred[t + s];
        __syncthreads();
    }
    if (t == 0) {
        out[LOSS_OFF] = 0.25f * (*lossAcc) * (1.0f / ((float)Nrows * CDdim));
        out[PERP_OFF] = expf(-sred[0]);
    }
}

// ---------------------------------------------------------------------------
// Host launcher
// ---------------------------------------------------------------------------
extern "C" void kernel_launch(void* const* d_in, const int* in_sizes, int n_in,
                              void* d_out, int out_size, void* d_ws, size_t ws_size,
                              hipStream_t stream) {
    const float* states      = (const float*)d_in[0];
    const float* next_states = (const float*)d_in[1];
    const float* W_s         = (const float*)d_in[2];
    const float* b_s         = (const float*)d_in[3];
    const float* W_p         = (const float*)d_in[4];
    const float* b_p         = (const float*)d_in[5];
    const float* codebook    = (const float*)d_in[6];
    float* out = (float*)d_out;

    char* ws = (char*)d_ws;
    size_t off = 0;
    auto carve = [&](size_t bytes) -> void* {
        void* p = ws + off;
        off = (off + bytes + 255) & ~(size_t)255;
        return p;
    };
    __bf16* Abf    = (__bf16*)carve((size_t)Mstack * Ddim * 2);        // 33.5 MB
    __bf16* Wspk   = (__bf16*)carve((size_t)Ddim * Edim * 2);
    __bf16* Wppk   = (__bf16*)carve((size_t)K2dim * CDdim * 2);
    __bf16* CBpk   = (__bf16*)carve((size_t)CDdim * NCdim * 2);
    __bf16* policy = (__bf16*)carve((size_t)Nrows * K2dim * 2);        // 67 MB
    float*  latf   = (float*)carve((size_t)Nrows * CDdim * 4);
    __bf16* latbf  = (__bf16*)carve((size_t)Nrows * CDdim * 2);
    float*  cn     = (float*)carve((size_t)NCdim * 4);
    int*    idxbuf = (int*)carve((size_t)Nrows * 4);
    int*    counts = (int*)carve((size_t)NCdim * 4);
    float*  lossAcc = (float*)carve(256);

    // 0) zero encodings / counts / loss accumulator
    init_kernel<<<4096, 256, 0, stream>>>(out + ENC_OFF, (size_t)Nrows * NCdim,
                                          counts, lossAcc);
    // 1) activations -> stacked bf16 A
    cvt_bf16_kernel<<<4096, 256, 0, stream>>>(states, Abf, (size_t)Nrows * Ddim / 4);
    cvt_bf16_kernel<<<4096, 256, 0, stream>>>(next_states, Abf + (size_t)Nrows * Ddim,
                                              (size_t)Nrows * Ddim / 4);
    // 2) pack weight matrices to bf16 in WMMA lane order
    pack_b_kernel<<<4096, 256, 0, stream>>>(W_s, Wspk, Ddim, Edim, 0);
    pack_b_kernel<<<4096, 256, 0, stream>>>(W_p, Wppk, K2dim, CDdim, 0);
    pack_b_kernel<<<1024, 256, 0, stream>>>(codebook, CBpk, CDdim, NCdim, 1);
    // 3) codebook norms
    cnorm_kernel<<<NCdim, 256, 0, stream>>>(codebook, cn);
    // 4) GEMM1: stacked encoder + gelu + concat (TDM-staged A)
    gemm1_kernel<<<dim3(Mstack / 64, Edim / 128), 256, 0, stream>>>(Abf, Wspk, b_s, policy);
    // 5) GEMM2: latent projection (TDM-staged A)
    gemm2_kernel<<<dim3(Nrows / 64, CDdim / 128), 256, 0, stream>>>(policy, Wppk, b_p,
                                                                    latf, latbf);
    // 6) GEMM3 + fused argmin over 1024 codes
    gemm3_argmin_kernel<<<Nrows / 16, 256, 0, stream>>>(latbf, CBpk, cn, idxbuf);
    // 7) gather / one-hot / histogram / loss accumulation
    scatter_kernel<<<Nrows, 256, 0, stream>>>(idxbuf, codebook, latf, out, counts, lossAcc);
    // 8) loss + perplexity
    finalize_kernel<<<1, 1024, 0, stream>>>(counts, lossAcc, out);
}
